// Model_61581241090200
// MI455X (gfx1250) — matrix operand order repaired and checked by reading, not compiled, not run on hardware
//
#include <hip/hip_runtime.h>
#include <hip/hip_bf16.h>

#define VOCAB  32000
#define HIDDEN 256
#define BATCH  32
#define SEQ    128
#define ROWS   (BATCH * SEQ)   // 4096
#define NTILES (VOCAB / 16)    // 2000

typedef __attribute__((ext_vector_type(16))) __bf16          v16bf;
typedef __attribute__((ext_vector_type(8)))  float           v8f;
typedef __attribute__((ext_vector_type(8)))  unsigned short  u16x8;
typedef __attribute__((ext_vector_type(16))) unsigned short  u16x16;

union BFrag { u16x16 u; v16bf b; };

__device__ __forceinline__ unsigned short f32_to_bf16_rne(float f) {
    unsigned int u = __float_as_uint(f);
    unsigned int r = u + 0x7FFFu + ((u >> 16) & 1u);
    return (unsigned short)(r >> 16);
}

__device__ __forceinline__ u16x8 ld8(const unsigned short* p) {
    return *reinterpret_cast<const u16x8*>(p);
}

__device__ __forceinline__ v16bf make_frag(u16x8 lo, u16x8 hi) {
    BFrag f;
    f.u = __builtin_shufflevector(lo, hi, 0,1,2,3,4,5,6,7,8,9,10,11,12,13,14,15);
    return f.b;
}

// ---------------------------------------------------------------------------
// f32 -> bf16 conversion (emb + weight matrices)
// ---------------------------------------------------------------------------
__global__ void cvt_bf16_kernel(const float* __restrict__ src,
                                unsigned short* __restrict__ dst, int n) {
    int i = blockIdx.x * blockDim.x + threadIdx.x;
    if (i < n) dst[i] = f32_to_bf16_rne(src[i]);
}

// ---------------------------------------------------------------------------
// x = emb[tokens] * mask  -> bf16  (one block per (b,t), 256 threads = H)
// ---------------------------------------------------------------------------
__global__ void embed_kernel(const int* __restrict__ tokens,
                             const int* __restrict__ lengths,
                             const float* __restrict__ emb,
                             unsigned short* __restrict__ xb) {
    const int bt = blockIdx.x;
    const int h  = threadIdx.x;
    const int b  = bt >> 7;        // SEQ = 128
    const int t  = bt & 127;
    const int tok = tokens[bt];
    const float mk = (t < lengths[b]) ? 1.0f : 0.0f;
    const float v  = emb[(size_t)tok * HIDDEN + h] * mk;
    xb[(size_t)bt * HIDDEN + h] = f32_to_bf16_rne(v);
}

// ---------------------------------------------------------------------------
// Parallel input projection: xproj = x @ Wx^T + b  ([ROWS,H] fp32).
// One block per 16-row M-tile, 128 threads = 4 waves, 4 N-tiles per wave.
// A fragments (x rows) cached in registers across all N-tiles.
// ---------------------------------------------------------------------------
__global__ __launch_bounds__(128, 1)
void xproj_kernel(const unsigned short* __restrict__ xb,   // [ROWS,H] bf16
                  const unsigned short* __restrict__ Wx,   // [H,H] bf16
                  const float* __restrict__ bias,          // [H]
                  float* __restrict__ xproj) {             // [ROWS,H]
    const int tid   = threadIdx.x;
    const int wave  = tid >> 5;
    const int lane  = tid & 31;
    const int l15   = lane & 15;
    const int koffA = (lane < 16) ? 0 : 8;
    const int koffB = (lane < 16) ? 0 : 8;
    const int mtile = blockIdx.x;
    const int rowA  = mtile * 16 + l15;
    const int rbase = (lane < 16) ? 0 : 8;

    v16bf afr[8];
    const unsigned short* arow = xb + (size_t)rowA * HIDDEN;
#pragma unroll
    for (int kt = 0; kt < 8; ++kt) {
        const int kb = kt * 32;
        afr[kt] = make_frag(ld8(arow + kb + koffA), ld8(arow + kb + koffA + 16));
    }

    for (int nt = wave; nt < HIDDEN / 16; nt += 4) {
        const int colB = nt * 16 + l15;
        const unsigned short* brow = Wx + (size_t)colB * HIDDEN + koffB;
        v8f acc = {};
#pragma unroll
        for (int kt = 0; kt < 8; ++kt) {
            v16bf bfr = make_frag(ld8(brow + kt * 32), ld8(brow + kt * 32 + 16));
            acc = __builtin_amdgcn_wmma_f32_16x16x32_bf16(false, afr[kt], false, bfr,
                                                          (short)0, acc, false, false);
        }
        const float bn = bias[colB];
#pragma unroll
        for (int r = 0; r < 8; ++r) {
            const int rowg = mtile * 16 + rbase + r;
            xproj[(size_t)rowg * HIDDEN + colB] = acc[r] + bn;
        }
    }
}

// ---------------------------------------------------------------------------
// Sequential scan: h_t = tanh(xproj_t + h @ Wh^T);  hs = h_t * mask.
// One persistent workgroup, 512 threads = 16 waves (4/SIMD -> ample VGPRs).
// Wave w owns output N-tile w and BOTH M-tiles (rows 0-15 and 16-31).
// Wh fragments register-cached across all 128 steps; h in LDS (16 KB).
// ---------------------------------------------------------------------------
__global__ __launch_bounds__(512, 1)
void rnn_scan_kernel(const float* __restrict__ xproj,           // [ROWS,H]
                     const unsigned short* __restrict__ Wh,     // [H,H] bf16
                     const int* __restrict__ lengths,           // [B]
                     float* __restrict__ hs_f32,                // [ROWS,H] or null
                     unsigned short* __restrict__ hs_bf16) {    // [ROWS,H]
    __shared__ unsigned short h_lds[BATCH * HIDDEN];  // 16 KB

    const int tid = threadIdx.x;
    for (int i = tid; i < BATCH * HIDDEN; i += 512) h_lds[i] = 0;
    __syncthreads();

    const int wave   = tid >> 5;                  // 0..15 == ntile
    const int lane   = tid & 31;
    const int l15    = lane & 15;
    const int koffA  = (lane < 16) ? 0 : 8;
    const int koffB  = (lane < 16) ? 0 : 8;
    const int colB   = wave * 16 + l15;
    const int rbase0 = (lane < 16) ? 0 : 8;       // rows for acc0 (mtile 0)
    const int rbase1 = 16 + rbase0;               // rows for acc1 (mtile 1)

    // register-cache Wh fragments for this wave's 16 output columns
    v16bf whf[8];
    const unsigned short* wrow = Wh + (size_t)colB * HIDDEN + koffB;
#pragma unroll
    for (int kt = 0; kt < 8; ++kt)
        whf[kt] = make_frag(ld8(wrow + kt * 32), ld8(wrow + kt * 32 + 16));

    int len0[8], len1[8];
#pragma unroll
    for (int r = 0; r < 8; ++r) {
        len0[r] = lengths[rbase0 + r];
        len1[r] = lengths[rbase1 + r];
    }

    for (int t = 0; t < SEQ; ++t) {
        v8f acc0, acc1;
#pragma unroll
        for (int r = 0; r < 8; ++r) {
            acc0[r] = xproj[((size_t)(rbase0 + r) * SEQ + t) * HIDDEN + colB];
            acc1[r] = xproj[((size_t)(rbase1 + r) * SEQ + t) * HIDDEN + colB];
        }
#pragma unroll
        for (int kt = 0; kt < 8; ++kt) {
            const int kb = kt * 32;
            v16bf ah0 = make_frag(ld8(h_lds + l15 * HIDDEN + kb + koffA),
                                  ld8(h_lds + l15 * HIDDEN + kb + koffA + 16));
            v16bf ah1 = make_frag(ld8(h_lds + (16 + l15) * HIDDEN + kb + koffA),
                                  ld8(h_lds + (16 + l15) * HIDDEN + kb + koffA + 16));
            acc0 = __builtin_amdgcn_wmma_f32_16x16x32_bf16(false, ah0, false, whf[kt],
                                                           (short)0, acc0, false, false);
            acc1 = __builtin_amdgcn_wmma_f32_16x16x32_bf16(false, ah1, false, whf[kt],
                                                           (short)0, acc1, false, false);
        }
        __syncthreads();   // all waves done reading old h
#pragma unroll
        for (int r = 0; r < 8; ++r) {
            const int m0 = rbase0 + r;
            const float hv0 = tanhf(acc0[r]);             // carried h UNmasked
            h_lds[m0 * HIDDEN + colB] = f32_to_bf16_rne(hv0);
            const float hm0 = (t < len0[r]) ? hv0 : 0.0f; // hs output masked
            const size_t o0 = ((size_t)m0 * SEQ + t) * HIDDEN + colB;
            hs_bf16[o0] = f32_to_bf16_rne(hm0);
            if (hs_f32) hs_f32[o0] = hm0;

            const int m1 = rbase1 + r;
            const float hv1 = tanhf(acc1[r]);
            h_lds[m1 * HIDDEN + colB] = f32_to_bf16_rne(hv1);
            const float hm1 = (t < len1[r]) ? hv1 : 0.0f;
            const size_t o1 = ((size_t)m1 * SEQ + t) * HIDDEN + colB;
            hs_bf16[o1] = f32_to_bf16_rne(hm1);
            if (hs_f32) hs_f32[o1] = hm1;
        }
        __syncthreads();   // new h visible before next step
    }
}

// ---------------------------------------------------------------------------
// logits = h2 @ emb^T, softmax, mask.
// 128 blocks x 256 threads; 32 rows (2 M-tiles) per block.  8 waves = 4
// N-groups x 2 M-tiles; the two waves of a group walk the SAME N-tile
// stream so their B (emb) loads are shared in the WGP$, halving L2 traffic.
// Pass 1 writes raw logits while tracking online (max,sum); pass 2 rescales.
// ---------------------------------------------------------------------------
__global__ __launch_bounds__(256, 1)
void proj_softmax_kernel(const unsigned short* __restrict__ h2b,  // [ROWS,H]
                         const unsigned short* __restrict__ embb, // [V,H]
                         const int* __restrict__ lengths,
                         float* __restrict__ probs) {             // [ROWS,V]
    __shared__ float smax[8][16];
    __shared__ float ssum[8][16];
    __shared__ float fmaxs[32];
    __shared__ float fscale[32];

    const int tid   = threadIdx.x;
    const int wave  = tid >> 5;          // 0..7
    const int g     = wave >> 1;         // N-group 0..3
    const int mt    = wave & 1;          // M-tile within block
    const int lane  = tid & 31;
    const int l15   = lane & 15;
    const int koffA = (lane < 16) ? 0 : 8;
    const int koffB = (lane < 16) ? 0 : 8;
    const int mtile = blockIdx.x * 2 + mt;
    const int rowA  = mtile * 16 + l15;
    const int rbase = (lane < 16) ? 0 : 8;

    // A fragments (h2 rows) register-resident for the whole N stream
    v16bf afr[8];
    const unsigned short* arow = h2b + (size_t)rowA * HIDDEN;
#pragma unroll
    for (int kt = 0; kt < 8; ++kt) {
        const int kb = kt * 32;
        afr[kt] = make_frag(ld8(arow + kb + koffA), ld8(arow + kb + koffA + 16));
    }

    float rmax[8], rsum[8];
#pragma unroll
    for (int r = 0; r < 8; ++r) { rmax[r] = -3.0e38f; rsum[r] = 0.0f; }

    for (int nt = g; nt < NTILES; nt += 4) {
        const int colB = nt * 16 + l15;
        const unsigned short* brow = embb + (size_t)colB * HIDDEN + koffB;
        if (nt + 4 < NTILES)   // global_prefetch_b8 of next tile's B operand
            __builtin_prefetch(brow + (size_t)64 * HIDDEN, 0, 3);
        v8f acc = {};
#pragma unroll
        for (int kt = 0; kt < 8; ++kt) {
            v16bf bfr = make_frag(ld8(brow + kt * 32), ld8(brow + kt * 32 + 16));
            acc = __builtin_amdgcn_wmma_f32_16x16x32_bf16(false, afr[kt], false, bfr,
                                                          (short)0, acc, false, false);
        }
#pragma unroll
        for (int r = 0; r < 8; ++r) {
            const float v = acc[r];
            const int rowg = mtile * 16 + rbase + r;
            probs[(size_t)rowg * VOCAB + colB] = v;       // raw logit, pass 1
            const float nm = fmaxf(rmax[r], v);           // online max/sum
            rsum[r] = rsum[r] * __expf(rmax[r] - nm) + __expf(v - nm);
            rmax[r] = nm;
        }
    }

    // reduce (max,sum) across the 16 lanes of each half-wave
#pragma unroll
    for (int off = 1; off < 16; off <<= 1) {
#pragma unroll
        for (int r = 0; r < 8; ++r) {
            const float om = __shfl_xor(rmax[r], off, 32);
            const float os = __shfl_xor(rsum[r], off, 32);
            const float nm = fmaxf(rmax[r], om);
            rsum[r] = rsum[r] * __expf(rmax[r] - nm) + os * __expf(om - nm);
            rmax[r] = nm;
        }
    }
    if (l15 == 0) {
#pragma unroll
        for (int r = 0; r < 8; ++r) {
            smax[wave][rbase + r] = rmax[r];
            ssum[wave][rbase + r] = rsum[r];
        }
    }
    __threadfence();      // make pass-1 logit stores visible to pass-2 readers
    __syncthreads();

    // merge the 4 waves (N-groups) that share each M-tile: waves {mt, mt+2, mt+4, mt+6}
    if (tid < 32) {
        const int mt2 = tid >> 4;        // which M-tile's rows
        const int rr  = tid & 15;
        float m = smax[mt2][rr], s = ssum[mt2][rr];
#pragma unroll
        for (int w = 1; w < 4; ++w) {
            const float om = smax[mt2 + 2 * w][rr];
            const float os = ssum[mt2 + 2 * w][rr];
            const float nm = fmaxf(m, om);
            s = s * __expf(m - nm) + os * __expf(om - nm);
            m = nm;
        }
        fmaxs[tid]  = m;
        fscale[tid] = (s > 0.0f) ? 1.0f / s : 0.0f;
    }
    __syncthreads();

    // pass 2: in-place exp/scale/mask over the block's 32 rows
    for (int r = 0; r < 32; ++r) {
        const int rowg = blockIdx.x * 32 + r;
        const int b = rowg >> 7;
        const int t = rowg & 127;
        const float mk    = (t < lengths[b]) ? 1.0f : 0.0f;
        const float scale = fscale[r] * mk;
        const float mx    = fmaxs[r];
        float* prow = probs + (size_t)rowg * VOCAB;
        for (int c = tid; c < VOCAB; c += 256)
            prow[c] = __expf(prow[c] - mx) * scale;
    }
}

// ---------------------------------------------------------------------------
extern "C" void kernel_launch(void* const* d_in, const int* in_sizes, int n_in,
                              void* d_out, int out_size, void* d_ws, size_t ws_size,
                              hipStream_t stream) {
    const int*   tokens  = (const int*)d_in[0];
    const int*   lengths = (const int*)d_in[1];
    const float* emb     = (const float*)d_in[2];
    const float* Wx1     = (const float*)d_in[3];
    const float* Wh1     = (const float*)d_in[4];
    const float* b1      = (const float*)d_in[5];
    const float* Wx2     = (const float*)d_in[6];
    const float* Wh2     = (const float*)d_in[7];
    const float* b2      = (const float*)d_in[8];

    float* probs  = (float*)d_out;                       // [ROWS, V]
    float* h2_f32 = probs + (size_t)ROWS * VOCAB;        // [ROWS, H]

    char* ws = (char*)d_ws;
    unsigned short* emb_b = (unsigned short*)ws;  ws += (size_t)VOCAB * HIDDEN * 2;
    unsigned short* wx1_b = (unsigned short*)ws;  ws += (size_t)HIDDEN * HIDDEN * 2;
    unsigned short* wh1_b = (unsigned short*)ws;  ws += (size_t)HIDDEN * HIDDEN * 2;
    unsigned short* wx2_b = (unsigned short*)ws;  ws += (size_t)HIDDEN * HIDDEN * 2;
    unsigned short* wh2_b = (unsigned short*)ws;  ws += (size_t)HIDDEN * HIDDEN * 2;
    unsigned short* x_b   = (unsigned short*)ws;  ws += (size_t)ROWS * HIDDEN * 2;
    unsigned short* h1_b  = (unsigned short*)ws;  ws += (size_t)ROWS * HIDDEN * 2;
    unsigned short* h2_b  = (unsigned short*)ws;  ws += (size_t)ROWS * HIDDEN * 2;
    float*          xp1   = (float*)ws;           ws += (size_t)ROWS * HIDDEN * 4;
    float*          xp2   = (float*)ws;           ws += (size_t)ROWS * HIDDEN * 4;

    const int nEmb = VOCAB * HIDDEN;
    const int nW   = HIDDEN * HIDDEN;
    cvt_bf16_kernel<<<(nEmb + 255) / 256, 256, 0, stream>>>(emb, emb_b, nEmb);
    cvt_bf16_kernel<<<(nW + 255) / 256, 256, 0, stream>>>(Wx1, wx1_b, nW);
    cvt_bf16_kernel<<<(nW + 255) / 256, 256, 0, stream>>>(Wh1, wh1_b, nW);
    cvt_bf16_kernel<<<(nW + 255) / 256, 256, 0, stream>>>(Wx2, wx2_b, nW);
    cvt_bf16_kernel<<<(nW + 255) / 256, 256, 0, stream>>>(Wh2, wh2_b, nW);

    embed_kernel<<<ROWS, HIDDEN, 0, stream>>>(tokens, lengths, emb, x_b);

    // layer 1: parallel input projection, then sequential scan
    xproj_kernel<<<ROWS / 16, 128, 0, stream>>>(x_b, wx1_b, b1, xp1);
    rnn_scan_kernel<<<1, 512, 0, stream>>>(xp1, wh1_b, lengths,
                                           (float*)nullptr, h1_b);

    // layer 2
    xproj_kernel<<<ROWS / 16, 128, 0, stream>>>(h1_b, wx2_b, b2, xp2);
    rnn_scan_kernel<<<1, 512, 0, stream>>>(xp2, wh2_b, lengths,
                                           h2_f32, h2_b);

    proj_softmax_kernel<<<ROWS / 32, 256, 0, stream>>>(h2_b, emb_b, lengths, probs);
}
